// AgentDecoderLSTM_12146167513572
// MI455X (gfx1250) — matrix operand order repaired and checked by reading, not compiled, not run on hardware
//
#include <hip/hip_runtime.h>

typedef __attribute__((ext_vector_type(16))) __bf16 v16bf;
typedef __attribute__((ext_vector_type(8)))  float  v8f;
typedef __attribute__((ext_vector_type(16))) unsigned short v16us;

union BFrag {
    v16bf bf;
    v16us us;
    uint4 q[2];
};

__device__ __forceinline__ unsigned short f32_to_bf16_bits(float f) {
    __bf16 b = (__bf16)f;                       // native v_cvt, RNE
    return __builtin_bit_cast(unsigned short, b);
}

#if __has_builtin(__builtin_amdgcn_tanhf)
__device__ __forceinline__ float fast_tanh(float x) {
    return __builtin_amdgcn_tanhf(x);           // v_tanh_f32 (1 trans op)
}
__device__ __forceinline__ float fast_sigmoid(float x) {
    return __builtin_fmaf(0.5f, __builtin_amdgcn_tanhf(0.5f * x), 0.5f);
}
#else
__device__ __forceinline__ float fast_sigmoid(float x) {
    return __builtin_amdgcn_rcpf(1.0f + __expf(-x));
}
__device__ __forceinline__ float fast_tanh(float x) {
    return 2.0f * fast_sigmoid(2.0f * x) - 1.0f;
}
#endif

// ---------------- prep kernel: fold embedding, convert weights to bf16 ----------
// ws layout (u16 elements):
//   [0,     4096)  whh_bf : W_hh rows, bf16
//   [4096,  4608)  whp_bf : W_hp padded to 16 output cols, bf16
//   [4608,  8704)  bp_bf  : per gate-row n: [Wps[n][0], Wps[n][1], btot[n], 0...]
//   then f32: bhp2[2]
__global__ void prep_kernel(const float* __restrict__ W_se, const float* __restrict__ b_se,
                            const float* __restrict__ W_ih, const float* __restrict__ b_ih,
                            const float* __restrict__ W_hh, const float* __restrict__ b_hh,
                            const float* __restrict__ W_hp, const float* __restrict__ b_hp,
                            unsigned short* __restrict__ whh_bf,
                            unsigned short* __restrict__ whp_bf,
                            unsigned short* __restrict__ bp_bf,
                            float* __restrict__ bhp2) {
    const int n = threadIdx.x;            // 0..127 (gate-output row)
    for (int k = 0; k < 32; ++k)
        whh_bf[n * 32 + k] = f32_to_bf16_bits(W_hh[n * 32 + k]);
    // folded rank-2 input path + total bias
    float s = b_ih[n] + b_hh[n];
    float p0 = 0.f, p1 = 0.f;
    for (int e = 0; e < 32; ++e) {
        float w = W_ih[n * 32 + e];
        s  += w * b_se[e];
        p0 += w * W_se[e * 2 + 0];
        p1 += w * W_se[e * 2 + 1];
    }
    bp_bf[n * 32 + 0] = f32_to_bf16_bits(p0);
    bp_bf[n * 32 + 1] = f32_to_bf16_bits(p1);
    bp_bf[n * 32 + 2] = f32_to_bf16_bits(s);
    for (int k = 3; k < 32; ++k) bp_bf[n * 32 + k] = 0;
    if (n < 16) {
        for (int k = 0; k < 32; ++k) {
            float v = (n < 2) ? W_hp[n * 32 + k] : 0.f;
            whp_bf[n * 32 + k] = f32_to_bf16_bits(v);
        }
    }
    if (n < 2) bhp2[n] = b_hp[n];
}

// ---------------- main LSTM kernel: 1 wave == 16 batch rows ---------------------
#define HSTRIDE 36   // floats; rows 16B-aligned, lanes hit disjoint 4-bank groups

__global__ __launch_bounds__(256)
__attribute__((amdgpu_num_vgpr(256)))
void lstm_kernel(
        const float* __restrict__ last_pos_rel,
        const float* __restrict__ hidden_state,
        const float* __restrict__ start_pos,
        const unsigned short* __restrict__ whh_bf,
        const unsigned short* __restrict__ whp_bf,
        const unsigned short* __restrict__ bp_bf,
        const float* __restrict__ bhp2,
        const int* __restrict__ seq_len_ptr,
        float* __restrict__ out, int B) {

    __shared__ float hstage[8][16 * HSTRIDE];       // per-wave h staging
    __shared__ float pstage[8][32];                 // per-wave pos_rel (16 rows x 2)
    __shared__ unsigned short wlds[128 * 32];       // W_hh  bf16, block-shared (8 KB)
    __shared__ unsigned short bplds[128 * 32];      // B_p   bf16, block-shared (8 KB)

    const int tid  = threadIdx.x;
    const int wave = tid >> 5;
    const int lane = tid & 31;
    const int n    = lane & 15;            // C-layout column / A-layout row
    const int half = lane >> 4;
    const int b0   = (blockIdx.x * 8 + wave) * 16;
    const int T    = *seq_len_ptr;

    float* hl = hstage[wave];
    float* pl = pstage[wave];

    // ---- cooperative copy of weight matrices into LDS (1024 uint4 total)
    {
        const uint4* sA = (const uint4*)whh_bf;
        const uint4* sB = (const uint4*)bp_bf;
        uint4* dA = (uint4*)wlds;
        uint4* dB = (uint4*)bplds;
        dA[tid]       = sA[tid];
        dA[tid + 256] = sA[tid + 256];
        dB[tid]       = sB[tid];
        dB[tid + 256] = sB[tid + 256];
    }

    // ---- loop-invariant small operands (registers)
    BFrag bhp;
    {
        const uint4* q = (const uint4*)(whp_bf + n * 32 + half * 16);
        bhp.q[0] = q[0];
        bhp.q[1] = q[1];
    }
    const float posbias = (n < 2) ? bhp2[n] : 0.0f;

    // ---- stage h0 and initial p into LDS
#pragma unroll
    for (int r = 0; r < 16; ++r)
        hl[r * HSTRIDE + lane] = hidden_state[(size_t)(b0 + r) * 32 + lane];
    pl[lane] = last_pos_rel[(size_t)b0 * 2 + lane];
    float cum = start_pos[(size_t)b0 * 2 + lane];

    v8f cc0 = {}, cc1 = {};        // cell state (f32, C-layout)
    v8f hc0 = {}, hc1 = {};        // hidden    (f32, C-layout)

    // A_h loader: contiguous-half 16-bit A layout (k = 16*half + s).
    BFrag ah;
    auto loadA = [&]() {
        const float4* row = (const float4*)(hl + n * HSTRIDE + half * 16);
        float4 f0 = row[0];
        float4 f1 = row[1];
        const float4* row2 = (const float4*)(hl + n * HSTRIDE + half * 16 + 8);
        float4 f2 = row2[0];
        float4 f3 = row2[1];
        ah.us[0]  = f32_to_bf16_bits(f0.x);  ah.us[1]  = f32_to_bf16_bits(f0.y);
        ah.us[2]  = f32_to_bf16_bits(f0.z);  ah.us[3]  = f32_to_bf16_bits(f0.w);
        ah.us[4]  = f32_to_bf16_bits(f1.x);  ah.us[5]  = f32_to_bf16_bits(f1.y);
        ah.us[6]  = f32_to_bf16_bits(f1.z);  ah.us[7]  = f32_to_bf16_bits(f1.w);
        ah.us[8]  = f32_to_bf16_bits(f2.x);  ah.us[9]  = f32_to_bf16_bits(f2.y);
        ah.us[10] = f32_to_bf16_bits(f2.z);  ah.us[11] = f32_to_bf16_bits(f2.w);
        ah.us[12] = f32_to_bf16_bits(f3.x);  ah.us[13] = f32_to_bf16_bits(f3.y);
        ah.us[14] = f32_to_bf16_bits(f3.z);  ah.us[15] = f32_to_bf16_bits(f3.w);
    };
    // A_p loader: row m holds [p0, p1, 1, 0...] over K; lanes 16-31 all zero.
    BFrag ap;
    auto loadAp = [&]() {
#pragma unroll
        for (int e = 0; e < 16; ++e) ap.us[e] = 0;
        if (half == 0) {
            ap.us[0] = f32_to_bf16_bits(pl[n * 2 + 0]);
            ap.us[1] = f32_to_bf16_bits(pl[n * 2 + 1]);
            ap.us[2] = (unsigned short)0x3F80;   // bf16 1.0
        }
    };
    loadA();    // h_{-1} = h0
    loadAp();   // p_{-1} = last_pos_rel

    __syncthreads();   // wlds/bplds ready for all waves

    // Opaque offset: empty asm "redefines" it each iteration so the weight
    // ds_loads cannot be hoisted out of the loop (and re-spilled to scratch).
    unsigned woff = 0;

    for (int t = 0; t < T; ++t) {
        asm volatile("" : "+s"(woff));
        const unsigned short* wbase = wlds  + woff;
        const unsigned short* pbase = bplds + woff;

        // gates[j] = (A_p x B_p[j])  then  += A_h x B_hh[j]   (all on matrix pipe)
        v8f cg[8];
#pragma unroll
        for (int j = 0; j < 8; ++j) {
            BFrag pj, bj;
            const uint4* qp = (const uint4*)(pbase + (j * 16 + n) * 32 + half * 16);
            pj.q[0] = qp[0];
            pj.q[1] = qp[1];
            const uint4* qh = (const uint4*)(wbase + (j * 16 + n) * 32 + half * 16);
            bj.q[0] = qh[0];
            bj.q[1] = qh[1];
            v8f zc = {};
            v8f ini = __builtin_amdgcn_wmma_f32_16x16x32_bf16(
                          false, ap.bf, false, pj.bf, (short)0, zc, false, false);
            cg[j] = __builtin_amdgcn_wmma_f32_16x16x32_bf16(
                        false, ah.bf, false, bj.bf, (short)0, ini, false, false);
        }
        // i,f,g,o elementwise (all same C-layout slot)
#pragma unroll
        for (int r = 0; r < 8; ++r) {
            float i0 = fast_sigmoid(cg[0][r]);
            float i1 = fast_sigmoid(cg[1][r]);
            float f0 = fast_sigmoid(cg[2][r]);
            float f1 = fast_sigmoid(cg[3][r]);
            float g0 = fast_tanh   (cg[4][r]);
            float g1 = fast_tanh   (cg[5][r]);
            float o0 = fast_sigmoid(cg[6][r]);
            float o1 = fast_sigmoid(cg[7][r]);
            cc0[r] = f0 * cc0[r] + i0 * g0;
            cc1[r] = f1 * cc1[r] + i1 * g1;
            hc0[r] = o0 * fast_tanh(cc0[r]);
            hc1[r] = o1 * fast_tanh(cc1[r]);
        }
        // h_t -> LDS (C-layout write, conflict-free)
#pragma unroll
        for (int r = 0; r < 8; ++r) {
            int m = half * 8 + r;
            hl[m * HSTRIDE + n]      = hc0[r];
            hl[m * HSTRIDE + 16 + n] = hc1[r];
        }
        loadA();   // A-layout h_t (per-wave LDS ops are in-order)

        // pos_rel = h_t @ W_hp_pad.T + b_hp  (one WMMA, cols 0..1 valid)
        v8f cp;
#pragma unroll
        for (int r = 0; r < 8; ++r) cp[r] = posbias;
        cp = __builtin_amdgcn_wmma_f32_16x16x32_bf16(
                 false, ah.bf, false, bhp.bf, (short)0, cp, false, false);

        if (n < 2) {
#pragma unroll
            for (int r = 0; r < 8; ++r)
                pl[(half * 8 + r) * 2 + n] = cp[r];
        }
        // cumsum + coalesced store: lane -> (row lane/2, comp lane%2)
        float pv = pl[lane];
        cum += pv;
        out[(size_t)t * B * 2 + (size_t)b0 * 2 + lane] = cum;

        loadAp();  // A_p for next step from fresh pos_rel
    }

    // hT output (second tuple element), concatenated after trajectory
    size_t hoff = (size_t)T * B * 2;
#pragma unroll
    for (int r = 0; r < 8; ++r) {
        int m = half * 8 + r;
        out[hoff + (size_t)(b0 + m) * 32 + n]      = hc0[r];
        out[hoff + (size_t)(b0 + m) * 32 + 16 + n] = hc1[r];
    }
}

extern "C" void kernel_launch(void* const* d_in, const int* in_sizes, int n_in,
                              void* d_out, int out_size, void* d_ws, size_t ws_size,
                              hipStream_t stream) {
    (void)n_in; (void)out_size; (void)ws_size;
    const float* last_pos_rel = (const float*)d_in[0];
    const float* hidden_state = (const float*)d_in[1];
    const float* start_pos    = (const float*)d_in[2];
    const float* W_se = (const float*)d_in[3];
    const float* b_se = (const float*)d_in[4];
    const float* W_ih = (const float*)d_in[5];
    const float* b_ih = (const float*)d_in[6];
    const float* W_hh = (const float*)d_in[7];
    const float* b_hh = (const float*)d_in[8];
    const float* W_hp = (const float*)d_in[9];
    const float* b_hp = (const float*)d_in[10];
    const int* seq_len = (const int*)d_in[11];

    const int B = in_sizes[0] / 2;

    unsigned short* whh_bf = (unsigned short*)d_ws;          // 4096 u16
    unsigned short* whp_bf = whh_bf + 128 * 32;              //  512 u16
    unsigned short* bp_bf  = whp_bf + 16 * 32;               // 4096 u16
    float* bhp2 = (float*)((char*)d_ws + (128 * 32 + 16 * 32 + 128 * 32) * sizeof(unsigned short));

    prep_kernel<<<1, 128, 0, stream>>>(W_se, b_se, W_ih, b_ih, W_hh, b_hh, W_hp, b_hp,
                                       whh_bf, whp_bf, bp_bf, bhp2);

    const int blocks = B / 128;   // 8 waves/block, 16 rows/wave
    lstm_kernel<<<blocks, 256, 0, stream>>>(last_pos_rel, hidden_state, start_pos,
                                            whh_bf, whp_bf, bp_bf, bhp2,
                                            seq_len, (float*)d_out, B);
}
